// RingAttention_37237366456727
// MI455X (gfx1250) — matrix operand order repaired and checked by reading, not compiled, not run on hardware
//
#include <hip/hip_runtime.h>
#include <hip/hip_bf16.h>
#include <math.h>

#define L_SEQ  4096
#define DMODEL 1024
#define HD     64
#define NH     16
#define NKV    4
#define KVD    256      // NKV*HD
#define BLK    1024

typedef __attribute__((ext_vector_type(16))) __bf16       v16bf;
typedef __attribute__((ext_vector_type(8)))  float        v8f;
typedef __attribute__((ext_vector_type(8)))  unsigned int v8u;

__device__ __forceinline__ unsigned short f2bf(float f) {
    unsigned int u = __builtin_bit_cast(unsigned int, f);
    unsigned int r = 0x7FFFu + ((u >> 16) & 1u);   // round-to-nearest-even
    return (unsigned short)((u + r) >> 16);
}

__device__ __forceinline__ v8f wmma_bf16(v16bf a, v16bf b, v8f c) {
    return __builtin_amdgcn_wmma_f32_16x16x32_bf16(false, a, false, b, (short)0, c, false, false);
}

// A-matrix (16x32 bf16) VGPR-j -> K index, contiguous order (k = 16g + 2j), matching B fills
__device__ __forceinline__ int a_kidx(int j, int g) {
    return g * 16 + 2 * j;
}

// async copy of 16 bytes global -> LDS, tracked by ASYNCcnt
__device__ __forceinline__ void async_b128(unsigned lds_off, const void* gptr) {
    unsigned long long ga = (unsigned long long)(uintptr_t)gptr;
    asm volatile("global_load_async_to_lds_b128 %0, %1, off"
                 :: "v"(lds_off), "v"(ga) : "memory");
}
__device__ __forceinline__ void wait_asynccnt0() {
    asm volatile("s_wait_asynccnt 0x0" ::: "memory");
}

// ---------------------------------------------------------------- convert
__global__ void cvt_bf16_kernel(const float* __restrict__ in,
                                unsigned short* __restrict__ out, int n) {
    int i = blockIdx.x * blockDim.x + threadIdx.x;
    if (i < n) out[i] = f2bf(in[i]);
}

// ---------------------------------------------------------------- GEMM: C = A (MxK) * W^T (W is NxK), f32 out
// Double-buffered LDS tiles staged with GLOBAL_LOAD_ASYNC_TO_LDS_B128.
__global__ __launch_bounds__(128)
void gemm_bf16_kernel(const unsigned short* __restrict__ A,
                      const unsigned short* __restrict__ W,
                      float* __restrict__ C, int M, int N, int K) {
    __shared__ unsigned short Abuf[2][64 * 32];   // 64 rows x 32 k (bf16)
    __shared__ unsigned short Bbuf[2][64 * 32];

    const int tid  = threadIdx.x;
    const int lane = tid & 31;
    const int wave = tid >> 5;
    const int n16  = lane & 15;
    const int g    = lane >> 4;
    const int wm   = wave >> 1, wn = wave & 1;
    const int m0g  = blockIdx.y * 64;
    const int n0g  = blockIdx.x * 64;

    v8f acc[2][2];
#pragma unroll
    for (int i = 0; i < 2; ++i)
#pragma unroll
        for (int j = 0; j < 2; ++j)
#pragma unroll
            for (int e = 0; e < 8; ++e) acc[i][j][e] = 0.0f;

    // stage one 64x32 A tile and one 64x32 B tile into LDS buffer `buf`
    auto stage = [&](int buf, int kb) {
#pragma unroll
        for (int i = 0; i < 2; ++i) {
            int c   = tid + i * 128;        // 0..255 16B-chunks per matrix
            int row = c >> 2;
            int seg = (c & 3) * 8;          // bf16 elements
            async_b128((unsigned)(uintptr_t)&Abuf[buf][row * 32 + seg],
                       A + (size_t)(m0g + row) * K + kb + seg);
            async_b128((unsigned)(uintptr_t)&Bbuf[buf][row * 32 + seg],
                       W + (size_t)(n0g + row) * K + kb + seg);
        }
    };

    stage(0, 0);
    wait_asynccnt0();
    __syncthreads();

    const int nk = K / 32;
    for (int t = 0; t < nk; ++t) {
        const int cur = t & 1;
        if (t + 1 < nk) stage(cur ^ 1, (t + 1) * 32);

        v8u au[2], bu[2];
#pragma unroll
        for (int i = 0; i < 2; ++i) {
            const unsigned short* ab = &Abuf[cur][(wm * 32 + i * 16 + n16) * 32];
#pragma unroll
            for (int j = 0; j < 8; ++j)
                au[i][j] = *(const unsigned int*)(ab + a_kidx(j, g));
            const unsigned short* bb = &Bbuf[cur][(wn * 32 + i * 16 + n16) * 32 + g * 16];
#pragma unroll
            for (int j = 0; j < 8; ++j)
                bu[i][j] = *(const unsigned int*)(bb + 2 * j);
        }
        v16bf av0 = __builtin_bit_cast(v16bf, au[0]);
        v16bf av1 = __builtin_bit_cast(v16bf, au[1]);
        v16bf bv0 = __builtin_bit_cast(v16bf, bu[0]);
        v16bf bv1 = __builtin_bit_cast(v16bf, bu[1]);
        acc[0][0] = wmma_bf16(av0, bv0, acc[0][0]);
        acc[0][1] = wmma_bf16(av0, bv1, acc[0][1]);
        acc[1][0] = wmma_bf16(av1, bv0, acc[1][0]);
        acc[1][1] = wmma_bf16(av1, bv1, acc[1][1]);

        if (t + 1 < nk) wait_asynccnt0();
        __syncthreads();
    }

#pragma unroll
    for (int i = 0; i < 2; ++i)
#pragma unroll
        for (int jn = 0; jn < 2; ++jn)
#pragma unroll
            for (int j = 0; j < 8; ++j)
                C[(size_t)(m0g + wm * 32 + i * 16 + j + 8 * g) * N +
                  n0g + wn * 32 + jn * 16 + n16] = acc[i][jn][j];
}

// ---------------------------------------------------------------- RoPE + pack (bf16 Q/K, bf16 V^T, repeated K/V outputs)
__global__ void rope_pack_kernel(const float* __restrict__ Qf,
                                 const float* __restrict__ Kf,
                                 const float* __restrict__ Vf,
                                 unsigned short* __restrict__ Qb,
                                 unsigned short* __restrict__ Kb,
                                 unsigned short* __restrict__ Vtb,
                                 float* __restrict__ k_out,
                                 float* __restrict__ v_out) {
    int idx = blockIdx.x * blockDim.x + threadIdx.x;
    const int NQ = L_SEQ * DMODEL;
    const int NK = L_SEQ * KVD;
    if (idx < NQ) {
        int l = idx / DMODEL, c = idx % DMODEL;
        int d = c & 63, i = d & 31;
        float freq = __powf(10000.0f, -(float)i * (1.0f / 32.0f));
        float ang = (float)l * freq;
        float cs = cosf(ang), sn = sinf(ang);
        float q  = Qf[idx];
        float qo = (d < 32) ? -Qf[idx + 32] : Qf[idx - 32];
        Qb[idx] = f2bf(q * cs + qo * sn);
    } else if (idx < NQ + NK) {
        int t = idx - NQ;
        int l = t / KVD, c = t % KVD;
        int d = c & 63, kvh = c >> 6, i = d & 31;
        float freq = __powf(10000.0f, -(float)i * (1.0f / 32.0f));
        float ang = (float)l * freq;
        float cs = cosf(ang), sn = sinf(ang);
        float k  = Kf[t];
        float ko = (d < 32) ? -Kf[t + 32] : Kf[t - 32];
        float kr = k * cs + ko * sn;
        Kb[t] = f2bf(kr);
#pragma unroll
        for (int rr = 0; rr < 4; ++rr)
            k_out[((size_t)(kvh * 4 + rr) * L_SEQ + l) * HD + d] = kr;
    } else if (idx < NQ + 2 * NK) {
        int t = idx - NQ - NK;
        int l = t / KVD, c = t % KVD;
        int d = c & 63, kvh = c >> 6;
        float v = Vf[t];
        Vtb[(size_t)c * L_SEQ + l] = f2bf(v);
#pragma unroll
        for (int rr = 0; rr < 4; ++rr)
            v_out[((size_t)(kvh * 4 + rr) * L_SEQ + l) * HD + d] = v;
    }
}

// ---------------------------------------------------------------- windowed flash attention (bf16 WMMA, f32 accum)
__global__ __launch_bounds__(128)
void attn_kernel(const unsigned short* __restrict__ Qb,
                 const unsigned short* __restrict__ Kb,
                 const unsigned short* __restrict__ Vtb,
                 unsigned short* __restrict__ attnb) {
    __shared__ unsigned short ldsP[4][16 * 32];   // per-wave 16x32 P tile (bf16)

    const int lane  = threadIdx.x & 31;
    const int wave  = threadIdx.x >> 5;
    const int n16   = lane & 15;
    const int g     = lane >> 4;
    const int head  = blockIdx.z;
    const int kvh   = head >> 2;
    const int qblk  = blockIdx.y;
    const int qrow0 = qblk * BLK + blockIdx.x * 64 + wave * 16;
    const float scale = 0.125f;   // 1/sqrt(64)

    // Q A-fragments: d in [0,32) and [32,64), fixed for whole key loop
    v8u qa0, qa1;
    {
        const unsigned short* qbase = Qb + (size_t)(qrow0 + n16) * DMODEL + head * HD;
#pragma unroll
        for (int j = 0; j < 8; ++j) {
            int k = a_kidx(j, g);
            qa0[j] = *(const unsigned int*)(qbase + k);
            qa1[j] = *(const unsigned int*)(qbase + 32 + k);
        }
    }
    v16bf qlo = __builtin_bit_cast(v16bf, qa0);
    v16bf qhi = __builtin_bit_cast(v16bf, qa1);

    float mrow[8], srow[8];
    v8f acc[4];
#pragma unroll
    for (int j = 0; j < 8; ++j) { mrow[j] = -1e30f; srow[j] = 0.0f; }
#pragma unroll
    for (int t = 0; t < 4; ++t)
#pragma unroll
        for (int e = 0; e < 8; ++e) acc[t][e] = 0.0f;

    const int nch = (qblk == 0) ? 32 : 96;   // 32-key chunks (prev block duplicated, per reference)
    for (int c = 0; c < nch; ++c) {
        int kb;
        if (qblk == 0)      kb = c * 32;
        else if (c < 64)    kb = (qblk - 1) * BLK + (c & 31) * 32;
        else                kb = qblk * BLK + (c - 64) * 32;

        // S tiles: two 16-key N tiles, each = two chained K=32 WMMAs over head dim
        v8f c0, c1;
#pragma unroll
        for (int t = 0; t < 2; ++t) {
            v8u bu0, bu1;
            const unsigned short* krow =
                Kb + (size_t)(kb + t * 16 + n16) * KVD + kvh * HD + g * 16;
#pragma unroll
            for (int j = 0; j < 8; ++j) {
                bu0[j] = *(const unsigned int*)(krow + 2 * j);        // d = g*16+2j
                bu1[j] = *(const unsigned int*)(krow + 32 + 2 * j);   // d = 32+g*16+2j
            }
            v8f z;
#pragma unroll
            for (int e = 0; e < 8; ++e) z[e] = 0.0f;
            v8f s = wmma_bf16(qlo, __builtin_bit_cast(v16bf, bu0), z);
            s     = wmma_bf16(qhi, __builtin_bit_cast(v16bf, bu1), s);
            if (t == 0) c0 = s; else c1 = s;
        }

        // online softmax update; P -> LDS (bf16, row-major 16x32)
#pragma unroll
        for (int j = 0; j < 8; ++j) {
            float s0 = c0[j] * scale, s1 = c1[j] * scale;
            float mx = fmaxf(s0, s1);
            mx = fmaxf(mx, __shfl_xor(mx, 1, 32));
            mx = fmaxf(mx, __shfl_xor(mx, 2, 32));
            mx = fmaxf(mx, __shfl_xor(mx, 4, 32));
            mx = fmaxf(mx, __shfl_xor(mx, 8, 32));
            float mnew = fmaxf(mrow[j], mx);
            float corr = __expf(mrow[j] - mnew);
            float p0 = __expf(s0 - mnew);
            float p1 = __expf(s1 - mnew);
            srow[j] = srow[j] * corr + p0 + p1;
            mrow[j] = mnew;
#pragma unroll
            for (int t = 0; t < 4; ++t) acc[t][j] *= corr;
            int r = j + 8 * g;
            ldsP[wave][r * 32 + n16]      = f2bf(p0);
            ldsP[wave][r * 32 + 16 + n16] = f2bf(p1);
        }
        __syncthreads();

        // reload P in A-matrix layout
        v8u pu;
#pragma unroll
        for (int j = 0; j < 8; ++j)
            pu[j] = *(const unsigned int*)&ldsP[wave][n16 * 32 + a_kidx(j, g)];
        v16bf pv = __builtin_bit_cast(v16bf, pu);
        __syncthreads();

        // O += P * V  (V^T storage makes key dim contiguous for B frags)
#pragma unroll
        for (int t = 0; t < 4; ++t) {
            v8u vu;
            const unsigned short* vrow =
                Vtb + (size_t)(kvh * HD + t * 16 + n16) * L_SEQ + kb + g * 16;
#pragma unroll
            for (int j = 0; j < 8; ++j)
                vu[j] = *(const unsigned int*)(vrow + 2 * j);
            acc[t] = wmma_bf16(pv, __builtin_bit_cast(v16bf, vu), acc[t]);
        }
    }

    // finalize: row-sum across the 16-lane half, normalize, emit bf16 attn output
#pragma unroll
    for (int j = 0; j < 8; ++j) {
        float s = srow[j];
        s += __shfl_xor(s, 1, 32);
        s += __shfl_xor(s, 2, 32);
        s += __shfl_xor(s, 4, 32);
        s += __shfl_xor(s, 8, 32);
        float inv = 1.0f / s;
        int r = qrow0 + j + 8 * g;
#pragma unroll
        for (int t = 0; t < 4; ++t)
            attnb[(size_t)r * DMODEL + head * HD + t * 16 + n16] = f2bf(acc[t][j] * inv);
    }
}

// ---------------------------------------------------------------- launch
extern "C" void kernel_launch(void* const* d_in, const int* in_sizes, int n_in,
                              void* d_out, int out_size, void* d_ws, size_t ws_size,
                              hipStream_t stream) {
    const float* x  = (const float*)d_in[0];
    const float* wq = (const float*)d_in[1];
    const float* wk = (const float*)d_in[2];
    const float* wv = (const float*)d_in[3];
    const float* wo = (const float*)d_in[4];

    float* out   = (float*)d_out;        // 4096*1024
    float* k_out = out + 4194304;        // (16,4096,64)
    float* v_out = out + 8388608;        // (16,4096,64)

    char* ws = (char*)d_ws;
    unsigned short* xb    = (unsigned short*)(ws + 0);          // 8,388,608 B
    unsigned short* wqb   = (unsigned short*)(ws + 8388608);    // 2,097,152 B
    unsigned short* wkb   = (unsigned short*)(ws + 10485760);   //   524,288 B
    unsigned short* wvb   = (unsigned short*)(ws + 11010048);   //   524,288 B
    unsigned short* wob   = (unsigned short*)(ws + 11534336);   // 2,097,152 B
    float*          Qf    = (float*)(ws + 13631488);            // 16,777,216 B
    float*          Kf    = (float*)(ws + 30408704);            //  4,194,304 B
    float*          Vf    = (float*)(ws + 34603008);            //  4,194,304 B
    unsigned short* Qb    = (unsigned short*)(ws + 38797312);   //  8,388,608 B
    unsigned short* Kb    = (unsigned short*)(ws + 47185920);   //  2,097,152 B
    unsigned short* Vtb   = (unsigned short*)(ws + 49283072);   //  2,097,152 B
    unsigned short* attnb = (unsigned short*)(ws + 51380224);   //  8,388,608 B

    // 1) f32 -> bf16 conversions
    cvt_bf16_kernel<<<4194304 / 256, 256, 0, stream>>>(x,  xb,  4194304);
    cvt_bf16_kernel<<<1048576 / 256, 256, 0, stream>>>(wq, wqb, 1048576);
    cvt_bf16_kernel<<< 262144 / 256, 256, 0, stream>>>(wk, wkb,  262144);
    cvt_bf16_kernel<<< 262144 / 256, 256, 0, stream>>>(wv, wvb,  262144);
    cvt_bf16_kernel<<<1048576 / 256, 256, 0, stream>>>(wo, wob, 1048576);

    // 2) QKV projections (WMMA bf16, f32 accumulate, async LDS staging)
    gemm_bf16_kernel<<<dim3(16, 64), 128, 0, stream>>>(xb, wqb, Qf, L_SEQ, DMODEL, DMODEL);
    gemm_bf16_kernel<<<dim3(4,  64), 128, 0, stream>>>(xb, wkb, Kf, L_SEQ, KVD,    DMODEL);
    gemm_bf16_kernel<<<dim3(4,  64), 128, 0, stream>>>(xb, wvb, Vf, L_SEQ, KVD,    DMODEL);

    // 3) RoPE + repacking + repeated K/V outputs
    rope_pack_kernel<<<6291456 / 256, 256, 0, stream>>>(Qf, Kf, Vf, Qb, Kb, Vtb, k_out, v_out);

    // 4) windowed flash attention
    attn_kernel<<<dim3(16, 4, 16), 128, 0, stream>>>(Qb, Kb, Vtb, attnb);

    // 5) output projection
    gemm_bf16_kernel<<<dim3(16, 64), 128, 0, stream>>>(attnb, wob, out, L_SEQ, DMODEL, DMODEL);
}
